// MyConv1D_4776003633326
// MI455X (gfx1250) — matrix operand order repaired and checked by reading, not compiled, not run on hardware
//
#include <hip/hip_runtime.h>

typedef __attribute__((ext_vector_type(16))) __bf16 v16bf;
typedef __attribute__((ext_vector_type(8)))  float   v8f;
typedef __attribute__((ext_vector_type(2)))  float   fl2;
typedef __attribute__((ext_vector_type(4)))  unsigned int u32x4;

#define B_   32
#define T_   4096
#define C_   128
#define F_   256
#define K_   5
#define R_   (K_ * C_)      // 640 reduction length
#define NCHUNK (R_ / 32)    // 20 chunks of K=32
#define MBLK 128            // output rows (t positions) per workgroup
#define XROWS (MBLK + K_ - 1)   // 132 staged input rows
#define XPITCH 136          // 128 + 8 bf16 pad -> 4-bank rotation per row

// ---------------------------------------------------------------------------
// Prep: split fp32 weights into bf16 hi/lo pair (w layout [F][K*C] == [f][r]).
// ---------------------------------------------------------------------------
__global__ void conv1d_wsplit_kernel(const float* __restrict__ w,
                                     __bf16* __restrict__ whi,
                                     __bf16* __restrict__ wlo) {
    int i = blockIdx.x * blockDim.x + threadIdx.x;
    if (i >= F_ * R_) return;
    float v  = w[i];
    __bf16 h = (__bf16)v;
    whi[i] = h;
    wlo[i] = (__bf16)(v - (float)h);
}

// ---------------------------------------------------------------------------
// Main conv-as-GEMM kernel, bf16x3 WMMA with fp32 accumulation.
// Block: 256 threads (8 waves). Tile: 128 (M=t) x 256 (N=f).
// Wave wn (0..7) owns the full 128 M rows x a 32-wide F slice:
//   8 M-tiles x 2 N-tiles = 16 accumulators of 16x16 fp32.
// x is staged ONCE per block into LDS, pre-split into bf16 hi/lo, so the
// inner loop is pure ds_load_b128 + global_load_b128 + v_wmma.
// ---------------------------------------------------------------------------
__global__ __launch_bounds__(256)
void conv1d_wmma_kernel(const float* __restrict__ x,
                        const __bf16* __restrict__ whi,
                        const __bf16* __restrict__ wlo,
                        const float* __restrict__ bias,
                        float* __restrict__ out) {
    __shared__ __bf16 xhi[XROWS * XPITCH];
    __shared__ __bf16 xlo[XROWS * XPITCH];

    const int tid  = threadIdx.x;
    const int bidx = blockIdx.x;
    const int b    = bidx >> 5;           // T_/MBLK = 32 tiles per batch row
    const int t0   = (bidx & 31) * MBLK;
    const float* xb = x + (size_t)b * T_ * C_;

    // ---- Stage + split x[t0-2 .. t0+129][0..127] into LDS bf16 hi/lo.
    // One float2 (pair of channels) per iteration -> packed b32 LDS stores.
    for (int p = tid; p < XROWS * (C_ / 2); p += 256) {
        int row = p >> 6;                 // 0..131
        int cp  = (p & 63) << 1;          // even channel
        int t   = t0 - 2 + row;
        fl2 v = {0.f, 0.f};
        if (t >= 0 && t < T_) {
            v = *(const fl2*)(xb + (size_t)t * C_ + cp);
        }
        __bf16 h0 = (__bf16)v.x;
        __bf16 h1 = (__bf16)v.y;
        __bf16 l0 = (__bf16)(v.x - (float)h0);
        __bf16 l1 = (__bf16)(v.y - (float)h1);
        union { __bf16 h[2]; unsigned u; } ph, pl;
        ph.h[0] = h0; ph.h[1] = h1;
        pl.h[0] = l0; pl.h[1] = l1;
        *(unsigned*)&xhi[row * XPITCH + cp] = ph.u;
        *(unsigned*)&xlo[row * XPITCH + cp] = pl.u;
    }
    __syncthreads();

    const int lane  = tid & 31;
    const int wn    = tid >> 5;           // wave id 0..7 -> F slice wn*32
    const int lhalf = lane >> 4;          // 0: lanes 0-15, 1: lanes 16-31
    const int l15   = lane & 15;

    v8f acc[8][2];
    {
        v8f z = {0.f, 0.f, 0.f, 0.f, 0.f, 0.f, 0.f, 0.f};
        #pragma unroll
        for (int mt = 0; mt < 8; ++mt)
            #pragma unroll
            for (int nt = 0; nt < 2; ++nt)
                acc[mt][nt] = z;
    }

    const u32x4* whi4 = (const u32x4*)whi;
    const u32x4* wlo4 = (const u32x4*)wlo;

    typedef union { u32x4 u[2]; v16bf v; } frag;

    for (int chunk = 0; chunk < NCHUNK; ++chunk) {
        const int kk    = chunk >> 2;             // conv tap 0..4
        const int cch   = chunk & 3;              // 32-wide channel chunk
        const int cbase = cch * 32 + lhalf * 16;  // per-lane channel base

        // ---- B fragments (32x16 bf16, K x N): col n = lane&15;
        // lanes 0-15 hold K=0..15, lanes 16-31 hold K=16..31.
        // Memory: whi[fcol*640 + chunk*32 + lhalf*16 + j], 32B contiguous.
        frag bh[2], bl[2];
        #pragma unroll
        for (int nt = 0; nt < 2; ++nt) {
            const int fcol  = wn * 32 + nt * 16 + l15;
            const int base4 = fcol * (R_ / 8) + chunk * 4 + lhalf * 2;
            bh[nt].u[0] = whi4[base4];
            bh[nt].u[1] = whi4[base4 + 1];
            bl[nt].u[0] = wlo4[base4];
            bl[nt].u[1] = wlo4[base4 + 1];
        }

        // ---- A fragments from pre-split LDS (contiguous K, matches B):
        // lane<16 : K=0..15 (row = lane); lane>=16: K=16..31.
        #pragma unroll
        for (int mt = 0; mt < 8; ++mt) {
            const int off = (mt * 16 + l15 + kk) * XPITCH + cbase;  // row <=131
            frag ah, al;
            ah.u[0] = *(const u32x4*)&xhi[off];
            ah.u[1] = *(const u32x4*)&xhi[off + 8];
            al.u[0] = *(const u32x4*)&xlo[off];
            al.u[1] = *(const u32x4*)&xlo[off + 8];
            #pragma unroll
            for (int nt = 0; nt < 2; ++nt) {
                acc[mt][nt] = __builtin_amdgcn_wmma_f32_16x16x32_bf16(
                    false, ah.v, false, bh[nt].v, (short)0, acc[mt][nt], false, false);
                acc[mt][nt] = __builtin_amdgcn_wmma_f32_16x16x32_bf16(
                    false, ah.v, false, bl[nt].v, (short)0, acc[mt][nt], false, false);
                acc[mt][nt] = __builtin_amdgcn_wmma_f32_16x16x32_bf16(
                    false, al.v, false, bh[nt].v, (short)0, acc[mt][nt], false, false);
            }
        }
    }

    // ---- Epilogue: C/D layout — VGPR g: row = g + 8*lhalf, col = lane&15.
    const size_t row_base = (size_t)b * T_ + t0;
    #pragma unroll
    for (int nt = 0; nt < 2; ++nt) {
        const int fcol = wn * 32 + nt * 16 + l15;
        const float bv = bias[fcol];
        #pragma unroll
        for (int mt = 0; mt < 8; ++mt) {
            const size_t r0 = row_base + mt * 16 + lhalf * 8;
            #pragma unroll
            for (int g = 0; g < 8; ++g) {
                out[(r0 + g) * F_ + fcol] = acc[mt][nt][g] + bv;
            }
        }
    }
}

// ---------------------------------------------------------------------------
extern "C" void kernel_launch(void* const* d_in, const int* in_sizes, int n_in,
                              void* d_out, int out_size, void* d_ws, size_t ws_size,
                              hipStream_t stream) {
    const float* x    = (const float*)d_in[0];   // [B,T,C]
    const float* w    = (const float*)d_in[1];   // [F,K,C]
    const float* bias = (const float*)d_in[2];   // [F]
    float* out        = (float*)d_out;           // [B,T,F]

    __bf16* whi = (__bf16*)d_ws;                          // F*R bf16
    __bf16* wlo = (__bf16*)((char*)d_ws + (size_t)F_ * R_ * sizeof(__bf16));

    conv1d_wsplit_kernel<<<(F_ * R_ + 255) / 256, 256, 0, stream>>>(w, whi, wlo);

    const int nblocks = B_ * (T_ / MBLK);   // 1024
    conv1d_wmma_kernel<<<nblocks, 256, 0, stream>>>(x, whi, wlo, bias, out);
}
